// TransductiveGAT_11793980195192
// MI455X (gfx1250) — matrix operand order repaired and checked by reading, not compiled, not run on hardware
//
#include <hip/hip_runtime.h>
#include <math.h>
#include <stdint.h>

typedef __attribute__((ext_vector_type(16))) _Float16 v16h;
typedef __attribute__((ext_vector_type(8)))  float    v8f;

#define NEG_SLOPE 0.2f

// ---------------------------------------------------------------------------
// Pack a [K x NCOLS] f32 weight matrix into CDNA5 WMMA B-fragment order, f16.
// Fragment (kt,nt) covers rows kt*32..+31, cols nt*16..+15.
// Per ISA 7.12.2 (16-bit B 32x16): lane = g*16+col, element j<8 -> K = g*8+j, j>=8 -> K = 16+g*8+j-8.
// Stored frag-major: F[((kt*ntiles+nt)*32 + lane)*16 + j]
// ---------------------------------------------------------------------------
__global__ void pack_w_kernel(const float* __restrict__ W, _Float16* __restrict__ F,
                              int K, int NCOLS) {
    int idx = blockIdx.x * blockDim.x + threadIdx.x;
    int total = K * NCOLS;
    if (idx >= total) return;
    int ntiles = NCOLS >> 4;
    int frag = idx >> 9;          // 512 halves per fragment (32 lanes * 16)
    int rem  = idx & 511;
    int lane = rem >> 4;
    int j    = rem & 15;
    int kt = frag / ntiles;
    int nt = frag % ntiles;
    int g   = lane >> 4;
    int col = lane & 15;
    int k = kt * 32 + (j < 8 ? g * 8 + j : 16 + g * 8 + (j - 8));
    F[idx] = (_Float16)W[(size_t)k * NCOLS + nt * 16 + col];
}

// ---------------------------------------------------------------------------
// H[N, NTILES*16] = X[N, KTOT] * W (pre-packed f16 B-fragments), f32 accumulate
// via v_wmma_f32_16x16x32_f16.
//  - Block = 256 threads = 8 waves, covers 128 rows x NTILES*16 cols.
//  - X tile (128 rows x 32 K, f32) staged in LDS, double buffered, using the
//    CDNA5 async DMA path: global_load_async_to_lds_b128 (ASYNCcnt) with
//    fully coalesced 128B segments, overlapped with WMMA on the other buffer.
//  - A-fragments built from LDS with f32->f16 cvt per the 16-bit A 16x32
//    layout: lane g*16+r holds row r; element j<8 -> K=g*8+j, j>=8 -> K=16+g*8+j-8.
//  - Also zero-initializes Agg at the same addresses (free init pass).
// ---------------------------------------------------------------------------
template<int NTILES, int KTOT>
__global__ __launch_bounds__(256) void gat_gemm_kernel(
    const float* __restrict__ X, const _Float16* __restrict__ Wf,
    float* __restrict__ H, float* __restrict__ Agg, int Nrows) {
    constexpr int KTILES = KTOT / 32;
    __shared__ float smf[2][128 * 32];            // 2 x 16KB double buffer

    const int tid  = threadIdx.x;
    const int wave = tid >> 5;
    const int lane = tid & 31;
    const int g = lane >> 4;
    const int r = lane & 15;
    const int rowbase_blk = blockIdx.x * 128;
    const int rowbase = rowbase_blk + wave * 16;
    const bool active = rowbase < Nrows;          // uniform per wave (N % 16 == 0)

    // Async-DMA one 128x32 f32 chunk into LDS buffer `b` (coalesced 128B/8-lane)
    auto load_chunk = [&](int kt, int b) {
#pragma unroll
        for (int i = 0; i < 4; ++i) {
            int linear = tid + i * 256;           // 0..1023 float4 segments
            int row = linear >> 3;
            int f4  = linear & 7;
            int rg = rowbase_blk + row;
            if (rg < Nrows) {
                const float* gp = X + (size_t)rg * KTOT + kt * 32 + f4 * 4;
                uint32_t ldsoff = (uint32_t)(uintptr_t)&smf[b][row * 32 + f4 * 4];
                asm volatile("global_load_async_to_lds_b128 %0, %1, off"
                             :: "v"(ldsoff), "v"((uint64_t)(uintptr_t)gp)
                             : "memory");
            }
        }
    };

    v8f acc[NTILES] = {};
    const v16h* Bf = (const v16h*)Wf;

    load_chunk(0, 0);
    asm volatile("s_wait_asynccnt 0" ::: "memory");
    __syncthreads();

    int buf = 0;
    for (int kt = 0; kt < KTILES; ++kt) {
        if (kt + 1 < KTILES) load_chunk(kt + 1, buf ^ 1);

        if (active) {
            const float* sp = &smf[buf][(wave * 16 + r) * 32];
            float4 f0 = *(const float4*)(sp + g * 8);
            float4 f1 = *(const float4*)(sp + g * 8 + 4);
            float4 f2 = *(const float4*)(sp + 16 + g * 8);
            float4 f3 = *(const float4*)(sp + 20 + g * 8);
            v16h a;
            a[0]=(_Float16)f0.x;  a[1]=(_Float16)f0.y;  a[2]=(_Float16)f0.z;  a[3]=(_Float16)f0.w;
            a[4]=(_Float16)f1.x;  a[5]=(_Float16)f1.y;  a[6]=(_Float16)f1.z;  a[7]=(_Float16)f1.w;
            a[8]=(_Float16)f2.x;  a[9]=(_Float16)f2.y;  a[10]=(_Float16)f2.z; a[11]=(_Float16)f2.w;
            a[12]=(_Float16)f3.x; a[13]=(_Float16)f3.y; a[14]=(_Float16)f3.z; a[15]=(_Float16)f3.w;
#pragma unroll
            for (int nt = 0; nt < NTILES; ++nt) {
                v16h b = Bf[(size_t)(kt * NTILES + nt) * 32 + lane];
                acc[nt] = __builtin_amdgcn_wmma_f32_16x16x32_f16(
                    false, a, false, b, (short)0, acc[nt], false, false);
            }
        }
        asm volatile("s_wait_asynccnt 0" ::: "memory");  // own DMA done
        __syncthreads();                                  // readers of buf done
        buf ^= 1;
    }

    if (!active) return;
    // C/D layout: VGPR i -> row rowbase + g*8 + i, col nt*16 + r
    constexpr int NCOLS = NTILES * 16;
#pragma unroll
    for (int nt = 0; nt < NTILES; ++nt) {
#pragma unroll
        for (int i = 0; i < 8; ++i) {
            int row = rowbase + g * 8 + i;
            size_t o = (size_t)row * NCOLS + nt * 16 + r;
            H[o]   = acc[nt][i];
            Agg[o] = 0.0f;
        }
    }
}

// ---------------------------------------------------------------------------
// Per-node attention logits:  AS[n,h] = sum_c H[n,h,c]*a_src[h,c] (same for AD)
// Also initializes segment-max table to -inf and denom to 0.
// ---------------------------------------------------------------------------
template<int HN, int C>
__global__ void alpha_init_kernel(const float* __restrict__ Hm,
                                  const float* __restrict__ Asrc, const float* __restrict__ Adst,
                                  float* __restrict__ AS, float* __restrict__ AD,
                                  float* __restrict__ M, float* __restrict__ DEN, int N) {
    int idx = blockIdx.x * blockDim.x + threadIdx.x;
    if (idx >= N * HN) return;
    int n = idx / HN, h = idx % HN;
    const float* hp = Hm + (size_t)n * HN * C + h * C;
    float s = 0.f, d = 0.f;
#pragma unroll
    for (int c = 0; c < C; ++c) {
        float v = hp[c];
        s += v * Asrc[h * C + c];
        d += v * Adst[h * C + c];
    }
    AS[idx] = s; AD[idx] = d;
    M[idx] = -INFINITY;
    DEN[idx] = 0.f;
}

// Monotone float atomic-max via int/uint punning (valid with -inf init).
__device__ __forceinline__ void atomicMaxF32(float* a, float v) {
    if (v >= 0.f) atomicMax((int*)a, __float_as_int(v));
    else          atomicMin((unsigned int*)a, __float_as_uint(v));
}

__device__ __forceinline__ void edge_endpoints(const int* __restrict__ ei, int E,
                                               int t, int& src, int& dst) {
    if (t < E) { src = ei[t]; dst = ei[E + t]; }
    else       { src = t - E; dst = t - E; }    // appended self loops
}

// Pass A: segment max of leaky_relu(as[src]+ad[dst]) over incoming edges.
template<int HN>
__global__ __launch_bounds__(256) void edge_max_kernel(
    const int* __restrict__ ei, int E, int N,
    const float* __restrict__ AS, const float* __restrict__ AD, float* __restrict__ M) {
    int t = blockIdx.x * blockDim.x + threadIdx.x;
    if (t >= E + N) return;
    int src, dst; edge_endpoints(ei, E, t, src, dst);
#pragma unroll
    for (int h = 0; h < HN; ++h) {
        float e = AS[src * HN + h] + AD[dst * HN + h];
        e = e > 0.f ? e : NEG_SLOPE * e;
        atomicMaxF32(&M[dst * HN + h], e);
    }
}

// Pass B: denom[dst,h] += exp(e - m[dst,h])
template<int HN>
__global__ __launch_bounds__(256) void edge_den_kernel(
    const int* __restrict__ ei, int E, int N,
    const float* __restrict__ AS, const float* __restrict__ AD,
    const float* __restrict__ M, float* __restrict__ DEN) {
    int t = blockIdx.x * blockDim.x + threadIdx.x;
    if (t >= E + N) return;
    int src, dst; edge_endpoints(ei, E, t, src, dst);
#pragma unroll
    for (int h = 0; h < HN; ++h) {
        float e = AS[src * HN + h] + AD[dst * HN + h];
        e = e > 0.f ? e : NEG_SLOPE * e;
        atomicAdd(&DEN[dst * HN + h], expf(e - M[dst * HN + h]));
    }
}

// Pass C: Agg[dst,h,:] += H[src,h,:] * alpha ; one thread per (edge, head).
template<int HN, int C>
__global__ __launch_bounds__(256) void edge_agg_kernel(
    const int* __restrict__ ei, int E, int N,
    const float* __restrict__ AS, const float* __restrict__ AD,
    const float* __restrict__ M, const float* __restrict__ DEN,
    const float* __restrict__ Hm, float* __restrict__ Agg) {
    int t = blockIdx.x * blockDim.x + threadIdx.x;
    if (t >= (E + N) * HN) return;
    int eidx = t / HN, h = t % HN;
    int src, dst; edge_endpoints(ei, E, eidx, src, dst);
    float e = AS[src * HN + h] + AD[dst * HN + h];
    e = e > 0.f ? e : NEG_SLOPE * e;
    float al = expf(e - M[dst * HN + h]) / (DEN[dst * HN + h] + 1e-16f);
    const float* hp = Hm  + (size_t)src * HN * C + h * C;
    float*       op = Agg + (size_t)dst * HN * C + h * C;
#pragma unroll
    for (int c = 0; c < C; ++c) atomicAdd(&op[c], hp[c] * al);
}

// In-place bias + ELU over [N, F]  (layer-1 output feeds GEMM2 directly)
__global__ void bias_elu_kernel(float* __restrict__ A, const float* __restrict__ B,
                                int N, int F) {
    int idx = blockIdx.x * blockDim.x + threadIdx.x;
    if (idx >= N * F) return;
    float v = A[idx] + B[idx % F];
    A[idx] = v > 0.f ? v : expf(v) - 1.f;
}

// out[n,c] = mean_h Agg[n,h,c] + b[c]
template<int HN, int C>
__global__ void mean_bias_kernel(const float* __restrict__ Agg, const float* __restrict__ B,
                                 float* __restrict__ O, int N) {
    int idx = blockIdx.x * blockDim.x + threadIdx.x;
    if (idx >= N * C) return;
    int n = idx / C, c = idx % C;
    float s = 0.f;
#pragma unroll
    for (int h = 0; h < HN; ++h) s += Agg[(size_t)n * HN * C + h * C + c];
    O[idx] = s * (1.0f / HN) + B[c];
}

// ---------------------------------------------------------------------------
extern "C" void kernel_launch(void* const* d_in, const int* in_sizes, int n_in,
                              void* d_out, int out_size, void* d_ws, size_t ws_size,
                              hipStream_t stream) {
    const float* x    = (const float*)d_in[0];
    const int*   ei   = (const int*)  d_in[1];
    const float* W1   = (const float*)d_in[2];
    const float* as1w = (const float*)d_in[3];
    const float* ad1w = (const float*)d_in[4];
    const float* b1   = (const float*)d_in[5];
    const float* W2   = (const float*)d_in[6];
    const float* as2w = (const float*)d_in[7];
    const float* ad2w = (const float*)d_in[8];
    const float* b2   = (const float*)d_in[9];
    float* out = (float*)d_out;

    const int F_IN = 512;
    const int N = in_sizes[0] / F_IN;      // 100000
    const int E = in_sizes[1] / 2;         // 3200000
    const int T = E + N;                   // edges incl. self loops

    // Workspace carve-out (256B aligned)
    char* p = (char*)d_ws;
    auto alloc = [&](size_t bytes) -> char* {
        char* r = p; p += (bytes + 255) & ~(size_t)255; return r;
    };
    _Float16* W1f  = (_Float16*)alloc((size_t)512 * 64  * sizeof(_Float16));
    _Float16* W2f  = (_Float16*)alloc((size_t)64  * 128 * sizeof(_Float16));
    float* h1   = (float*)alloc((size_t)N * 64  * 4);
    float* agg1 = (float*)alloc((size_t)N * 64  * 4);
    float* as1  = (float*)alloc((size_t)N * 8 * 4);
    float* ad1  = (float*)alloc((size_t)N * 8 * 4);
    float* m1   = (float*)alloc((size_t)N * 8 * 4);
    float* den1 = (float*)alloc((size_t)N * 8 * 4);
    float* h2   = (float*)alloc((size_t)N * 128 * 4);
    float* agg2 = (float*)alloc((size_t)N * 128 * 4);
    float* as2  = (float*)alloc((size_t)N * 8 * 4);
    float* ad2  = (float*)alloc((size_t)N * 8 * 4);
    float* m2   = (float*)alloc((size_t)N * 8 * 4);
    float* den2 = (float*)alloc((size_t)N * 8 * 4);

    auto cdiv = [](int a, int b) { return (a + b - 1) / b; };
    const int BT = 256;

    // Pack weights into WMMA B-fragment layout (f16)
    pack_w_kernel<<<cdiv(512 * 64, BT),  BT, 0, stream>>>(W1, W1f, 512, 64);
    pack_w_kernel<<<cdiv(64 * 128, BT),  BT, 0, stream>>>(W2, W2f, 64, 128);

    // ---- Layer 1 ----
    int gemmBlocks = cdiv(N, 128);
    gat_gemm_kernel<4, 512><<<gemmBlocks, BT, 0, stream>>>(x, W1f, h1, agg1, N);
    alpha_init_kernel<8, 8><<<cdiv(N * 8, BT), BT, 0, stream>>>(
        h1, as1w, ad1w, as1, ad1, m1, den1, N);
    edge_max_kernel<8><<<cdiv(T, BT), BT, 0, stream>>>(ei, E, N, as1, ad1, m1);
    edge_den_kernel<8><<<cdiv(T, BT), BT, 0, stream>>>(ei, E, N, as1, ad1, m1, den1);
    edge_agg_kernel<8, 8><<<cdiv(T * 8, BT), BT, 0, stream>>>(
        ei, E, N, as1, ad1, m1, den1, h1, agg1);
    bias_elu_kernel<<<cdiv(N * 64, BT), BT, 0, stream>>>(agg1, b1, N, 64);

    // ---- Layer 2 ---- (agg1 now holds elu(conv1 out + b1))
    gat_gemm_kernel<8, 64><<<gemmBlocks, BT, 0, stream>>>(agg1, W2f, h2, agg2, N);
    alpha_init_kernel<8, 16><<<cdiv(N * 8, BT), BT, 0, stream>>>(
        h2, as2w, ad2w, as2, ad2, m2, den2, N);
    edge_max_kernel<8><<<cdiv(T, BT), BT, 0, stream>>>(ei, E, N, as2, ad2, m2);
    edge_den_kernel<8><<<cdiv(T, BT), BT, 0, stream>>>(ei, E, N, as2, ad2, m2, den2);
    edge_agg_kernel<8, 16><<<cdiv(T * 8, BT), BT, 0, stream>>>(
        ei, E, N, as2, ad2, m2, den2, h2, agg2);
    mean_bias_kernel<8, 16><<<cdiv(N * 16, BT), BT, 0, stream>>>(agg2, b2, out, N);
}
